// WaveQuantumTransformerLayer_46359876993381
// MI455X (gfx1250) — matrix operand order repaired and checked by reading, not compiled, not run on hardware
//
#include <hip/hip_runtime.h>

// ---------------------------------------------------------------------------
// WaveQuantumTransformerLayer for MI455X (gfx1250, wave32, WMMA bf16 path)
// ---------------------------------------------------------------------------
typedef __bf16 bf16;
typedef __attribute__((ext_vector_type(16))) __bf16 v16bf;
typedef __attribute__((ext_vector_type(8)))  __bf16 v8bf;
typedef __attribute__((ext_vector_type(8)))  float  v8f;
typedef int v4i __attribute__((vector_size(16)));   // matches builtin param type

#define B_   4
#define L_   1024
#define D_   1024
#define H_   16
#define DH_  64
#define BLD_ (B_ * L_ * D_)      // 4194304
#define DT_  0.1f
#define EPS_ 1e-5f

// ----- CDNA5 async Global->LDS path (guarded; fallback = sync VGPR staging) -
// Signature (from hipcc diagnostic): (int4 AS1*, int4 AS3*, imm offset, imm cpol)
#if __has_builtin(__builtin_amdgcn_global_load_async_to_lds_b128)
#define HAVE_ASYNC_LDS 1
#define AS1 __attribute__((address_space(1)))
#define AS3 __attribute__((address_space(3)))
#define ASYNC_B128(G, L) \
  __builtin_amdgcn_global_load_async_to_lds_b128((AS1 v4i*)(G), (AS3 v4i*)(L), 0, 0)
#else
#define HAVE_ASYNC_LDS 0
#endif

#if __has_builtin(__builtin_amdgcn_s_wait_asynccnt)
#define WAIT_ASYNC(N) __builtin_amdgcn_s_wait_asynccnt(N)
#else
#define WAIT_ASYNC(N) asm volatile("s_wait_asynccnt %0" ::"i"(N) : "memory")
#endif

static __device__ __forceinline__ v8f zero8() {
  v8f z = {0.f, 0.f, 0.f, 0.f, 0.f, 0.f, 0.f, 0.f};
  return z;
}

#define WMMA_BF16(A, Bm, C) \
  __builtin_amdgcn_wmma_f32_16x16x32_bf16(false, (A), false, (Bm), (short)0, (C), false, false)

// A-matrix fragment (16x32 bf16, MxK): lane m = lane&15; lanes<16 hold K 0-15,
// lanes>=16 hold K 16-31 (matches frag_b).  Two 16B contiguous loads.
static __device__ __forceinline__ v16bf frag_a(const bf16* p, int ld, int lane) {
  const int m  = lane & 15;
  const int kb = (lane >> 4) << 4;          // 0 or 16
  const bf16* q = p + m * ld + kb;
  v8bf lo = *(const v8bf*)(q);
  v8bf hi = *(const v8bf*)(q + 8);
  return __builtin_shufflevector(lo, hi, 0,1,2,3,4,5,6,7,8,9,10,11,12,13,14,15);
}

// B-matrix fragment (32x16 bf16, KxN) sourced from the row-major [N][K] matrix
// (i.e. W rows for x@W^T): lane n = lane&15; lanes<16 hold K 0-15, lanes>=16
// hold K 16-31 -> one contiguous 32B load per lane.
static __device__ __forceinline__ v16bf frag_b(const bf16* p, int ld, int lane) {
  const int n  = lane & 15;
  const int kb = (lane >> 4) << 4;          // 0 or 16
  return *(const v16bf*)(p + n * ld + kb);
}

// ---------------------------------------------------------------------------
// conversions
// ---------------------------------------------------------------------------
__global__ __launch_bounds__(256) void conv_weights_kernel(
    const float* __restrict__ Wq, const float* __restrict__ Wk,
    const float* __restrict__ Wv, const float* __restrict__ Wo,
    const float* __restrict__ Wsem, const float* __restrict__ Wsyn,
    const float* __restrict__ Wrea,
    bf16* __restrict__ wq, bf16* __restrict__ wk, bf16* __restrict__ wv,
    bf16* __restrict__ wo, bf16* __restrict__ wh) {
  const int i = blockIdx.x * 256 + threadIdx.x;
  wq[i] = (bf16)Wq[i];
  wk[i] = (bf16)Wk[i];
  wv[i] = (bf16)Wv[i];
  wo[i] = (bf16)Wo[i];
  wh[i] = (bf16)(Wsem[i] + Wsyn[i] + Wrea[i]);
}

__global__ __launch_bounds__(256) void conv_inputs_kernel(
    const float* __restrict__ r, const float* __restrict__ im,
    bf16* __restrict__ xr, bf16* __restrict__ xi) {
  const size_t i = (size_t)blockIdx.x * 256 + threadIdx.x;
  xr[i] = (bf16)r[i];
  xi[i] = (bf16)im[i];
}

// ---------------------------------------------------------------------------
// Tiled WMMA GEMM:  out[m,n] = sum_k A[m,k] * W[n,k]  (+ bias[n])
// BM=128, BN=64, BK=64, 256 threads = 8 waves (4x2), 32x32 per wave,
// 8 WMMAs per wave per K stage.  Async Global->LDS double-buffered pipeline
// when the CDNA5 builtin is available (ASYNCcnt-tracked, in-order, so
// s_wait_asynccnt <= 6 after issuing the next stage's 6 ops/wave guarantees
// the current stage has landed); otherwise single-buffer sync staging.
// outF: fp32 [M,N] (optional).  outH: bf16 head-major [B,H,L,Dh] (optional,
// only for M=4096,N=1024) so attention can fragment-load straight from global.
// ---------------------------------------------------------------------------
__global__ __launch_bounds__(256) void gemm_xwT_bf16(
    const bf16* __restrict__ A, const bf16* __restrict__ W,
    const float* __restrict__ bias,
    float* __restrict__ outF, bf16* __restrict__ outH,
    int M, int N, int K) {
  constexpr int NBUF = HAVE_ASYNC_LDS ? 2 : 1;
  __shared__ bf16 As[NBUF][128][64];   // 16 KB per buffer
  __shared__ bf16 Bs[NBUF][64][64];    //  8 KB per buffer

  const int tid  = threadIdx.x;
  const int lane = tid & 31;
  const int wave = tid >> 5;
  const int wm   = wave >> 1;        // 0..3 : 32-row strip
  const int wn   = wave & 1;         // 0..1 : 32-col strip
  const int m0   = blockIdx.y * 128;
  const int n0   = blockIdx.x * 64;

  v8f acc00 = zero8(), acc01 = zero8(), acc10 = zero8(), acc11 = zero8();

  // per-thread tile-load assignment
  const int lar = tid >> 1;                 // A row (0..127)
  const int lac = (tid & 1) * 32;           // A col (0 or 32)
  const int lbr = tid >> 2;                 // B row (0..63)
  const int lbc = (tid & 3) * 16;           // B col (0,16,32,48)

  const int nk = K >> 6;                    // K stages of 64

#define COMPUTE(BUF)                                                      \
  do {                                                                    \
    v16bf a00 = frag_a(&As[BUF][wm * 32 +  0][0],  64, lane);             \
    v16bf a01 = frag_a(&As[BUF][wm * 32 +  0][32], 64, lane);             \
    v16bf a10 = frag_a(&As[BUF][wm * 32 + 16][0],  64, lane);             \
    v16bf a11 = frag_a(&As[BUF][wm * 32 + 16][32], 64, lane);             \
    v16bf b00 = frag_b(&Bs[BUF][wn * 32 +  0][0],  64, lane);             \
    v16bf b01 = frag_b(&Bs[BUF][wn * 32 +  0][32], 64, lane);             \
    v16bf b10 = frag_b(&Bs[BUF][wn * 32 + 16][0],  64, lane);             \
    v16bf b11 = frag_b(&Bs[BUF][wn * 32 + 16][32], 64, lane);             \
    acc00 = WMMA_BF16(a00, b00, acc00);                                   \
    acc00 = WMMA_BF16(a01, b01, acc00);                                   \
    acc01 = WMMA_BF16(a00, b10, acc01);                                   \
    acc01 = WMMA_BF16(a01, b11, acc01);                                   \
    acc10 = WMMA_BF16(a10, b00, acc10);                                   \
    acc10 = WMMA_BF16(a11, b01, acc10);                                   \
    acc11 = WMMA_BF16(a10, b10, acc11);                                   \
    acc11 = WMMA_BF16(a11, b11, acc11);                                   \
  } while (0)

#if HAVE_ASYNC_LDS
  // 6 async B128s per thread per stage: 4 for the A tile half-row (64B),
  // 2 for the B tile quarter-row (32B).
#define ISSUE(KT, BUF)                                                    \
  do {                                                                    \
    const bf16* ga = A + (size_t)(m0 + lar) * K + (KT) * 64 + lac;        \
    bf16* la = &As[BUF][lar][lac];                                        \
    ASYNC_B128(ga +  0, la +  0);                                         \
    ASYNC_B128(ga +  8, la +  8);                                         \
    ASYNC_B128(ga + 16, la + 16);                                         \
    ASYNC_B128(ga + 24, la + 24);                                         \
    const bf16* gb = W + (size_t)(n0 + lbr) * K + (KT) * 64 + lbc;        \
    bf16* lb = &Bs[BUF][lbr][lbc];                                        \
    ASYNC_B128(gb + 0, lb + 0);                                           \
    ASYNC_B128(gb + 8, lb + 8);                                           \
  } while (0)

  ISSUE(0, 0);
  for (int kt = 0; kt < nk; ++kt) {
    const int cur = kt & 1;
    if (kt + 1 < nk) {
      ISSUE(kt + 1, cur ^ 1);
      WAIT_ASYNC(6);                 // in-order: previous stage complete
    } else {
      WAIT_ASYNC(0);
    }
    __syncthreads();                 // stage `cur` visible to all waves
    COMPUTE(cur);
    __syncthreads();                 // all reads done before buffer reuse
  }
#undef ISSUE
#else
  for (int kt = 0; kt < nk; ++kt) {
    const int k0 = kt * 64;
    *(v16bf*)&As[0][lar][lac] =
        *(const v16bf*)(A + (size_t)(m0 + lar) * K + k0 + lac);
    *(v16bf*)&As[0][lar][lac + 16] =
        *(const v16bf*)(A + (size_t)(m0 + lar) * K + k0 + lac + 16);
    *(v16bf*)&Bs[0][lbr][lbc] =
        *(const v16bf*)(W + (size_t)(n0 + lbr) * K + k0 + lbc);
    if (k0 + 64 < K)
      __builtin_prefetch(A + (size_t)(m0 + lar) * K + k0 + 64 + lac, 0, 3);
    __syncthreads();
    COMPUTE(0);
    __syncthreads();
  }
#endif
#undef COMPUTE

  // C/D layout: VGPR j -> M = j + (lane>=16)*8 ; N = lane&15
  const int mloc = (lane >> 4) << 3;
  const int nloc = lane & 15;

#define EPILOGUE(ACC, FM, FN)                                                  \
  do {                                                                         \
    const int mb  = m0 + wm * 32 + (FM) * 16 + mloc;                           \
    const int nb  = n0 + wn * 32 + (FN) * 16 + nloc;                           \
    const float bia = bias ? bias[nb] : 0.f;                                   \
    for (int j = 0; j < 8; ++j) {                                              \
      const float v  = (ACC)[j] + bia;                                         \
      const int   mm = mb + j;                                                 \
      if (outF) outF[(size_t)mm * N + nb] = v;                                 \
      if (outH) {                                                              \
        const int bb2 = mm >> 10, ll2 = mm & 1023;                             \
        const int hh2 = nb >> 6, dd2 = nb & 63;                                \
        outH[(((size_t)bb2 * H_ + hh2) * L_ + ll2) * DH_ + dd2] = (bf16)v;     \
      }                                                                        \
    }                                                                          \
  } while (0)

  EPILOGUE(acc00, 0, 0);
  EPILOGUE(acc01, 0, 1);
  EPILOGUE(acc10, 1, 0);
  EPILOGUE(acc11, 1, 1);
#undef EPILOGUE
}

// ---------------------------------------------------------------------------
// Fused attention: per block = one (b,h) and 16 q-rows.
//   scores (re,im) [16][1024] f32 built in LDS via 8 WMMAs per 16-key tile,
//   then per-row softmax of (re^2+im^2)/8 + phase sums using
//   cosφ = re/|z|, sinφ = im/|z|  (no atan2/cos/sin).
// LDS: 2*16*1024*4 = 128 KB of the 320 KB WGP LDS.
// ---------------------------------------------------------------------------
__global__ __launch_bounds__(256) void attn_kernel(
    const bf16* __restrict__ Qr, const bf16* __restrict__ Qi,
    const bf16* __restrict__ Kr, const bf16* __restrict__ Ki,
    float* __restrict__ att, float* __restrict__ Cb, float* __restrict__ Sb) {
  extern __shared__ float sm[];
  float* Sre = sm;              // [16][1024]
  float* Sim = sm + 16 * 1024;  // [16][1024]

  const int lane = threadIdx.x & 31;
  const int wave = threadIdx.x >> 5;
  const int qt = blockIdx.x & 63;  // q tile within head
  const int bh = blockIdx.x >> 6;  // b*H + h
  const size_t hoff = (size_t)bh * L_ * DH_;

  const bf16* qr = Qr + hoff + (size_t)(qt * 16) * DH_;
  const bf16* qi = Qi + hoff + (size_t)(qt * 16) * DH_;
  const v16bf aQr0 = frag_a(qr, DH_, lane), aQr1 = frag_a(qr + 32, DH_, lane);
  const v16bf aQi0 = frag_a(qi, DH_, lane), aQi1 = frag_a(qi + 32, DH_, lane);

  for (int kt = wave; kt < 64; kt += 8) {
    const bf16* kr = Kr + hoff + (size_t)(kt * 16) * DH_;
    const bf16* ki = Ki + hoff + (size_t)(kt * 16) * DH_;
    const v16bf bKr0 = frag_b(kr, DH_, lane), bKr1 = frag_b(kr + 32, DH_, lane);
    const v16bf bKi0 = frag_b(ki, DH_, lane), bKi1 = frag_b(ki + 32, DH_, lane);

    v8f re = zero8();
    re = WMMA_BF16(aQr0, bKr0, re);
    re = WMMA_BF16(aQr1, bKr1, re);
    re = WMMA_BF16(aQi0, bKi0, re);
    re = WMMA_BF16(aQi1, bKi1, re);
    v8f p1 = zero8();
    p1 = WMMA_BF16(aQr0, bKi0, p1);
    p1 = WMMA_BF16(aQr1, bKi1, p1);
    v8f p2 = zero8();
    p2 = WMMA_BF16(aQi0, bKr0, p2);
    p2 = WMMA_BF16(aQi1, bKr1, p2);
    const v8f im = p1 - p2;

    const int n   = (lane & 15) + kt * 16;
    const int mhi = (lane >> 4) << 3;
    for (int j = 0; j < 8; ++j) {
      Sre[(mhi + j) * 1024 + n] = re[j];
      Sim[(mhi + j) * 1024 + n] = im[j];
    }
  }
  __syncthreads();

  // softmax + C/S: each wave owns two of the 16 rows
  for (int rep = 0; rep < 2; ++rep) {
    const int row = wave * 2 + rep;
    float* sr = Sre + row * 1024;
    float* si = Sim + row * 1024;

    float mx = -3.0e38f;
    for (int k = lane; k < 1024; k += 32) {
      const float x = sr[k], y = si[k];
      mx = fmaxf(mx, (x * x + y * y) * 0.125f);     // /sqrt(Dh)=*0.125
    }
    for (int o = 16; o; o >>= 1) mx = fmaxf(mx, __shfl_xor(mx, o, 32));

    float sum = 0.f, cr = 0.f, ci = 0.f;
    for (int k = lane; k < 1024; k += 32) {
      const float x  = sr[k], y = si[k];
      const float m2 = x * x + y * y;
      const float e  = __expf(m2 * 0.125f - mx);
      const float rm = (m2 > 0.f) ? rsqrtf(m2) : 0.f;
      const float cv = (m2 > 0.f) ? x * rm : 1.f;   // cos(atan2(0,0)) = 1
      const float sv = y * rm;
      sum += e;
      cr += e * cv;
      ci += e * sv;
      sr[k] = e;                                    // stash unnormalized p
    }
    for (int o = 16; o; o >>= 1) {
      sum += __shfl_xor(sum, o, 32);
      cr  += __shfl_xor(cr, o, 32);
      ci  += __shfl_xor(ci, o, 32);
    }
    const float inv = 1.f / sum;
    const size_t abase = ((size_t)bh * L_ + qt * 16 + row) * (size_t)L_;
    for (int k = lane; k < 1024; k += 32) att[abase + k] = sr[k] * inv;
    if (lane == 0) {
      Cb[(size_t)bh * L_ + qt * 16 + row] = cr * inv;
      Sb[(size_t)bh * L_ + qt * 16 + row] = ci * inv;
    }
  }
}

// ---------------------------------------------------------------------------
// out_r = Vr*C - Vi*S ; out_i = Vr*S + Vi*C   (per q; factorized rotation)
// Emits bf16 merged [B*L, D] ready for the Wo GEMM.
// ---------------------------------------------------------------------------
__global__ __launch_bounds__(256) void rotate_kernel(
    const float* __restrict__ Vr, const float* __restrict__ Vi,
    const float* __restrict__ Cb, const float* __restrict__ Sb,
    bf16* __restrict__ Mr, bf16* __restrict__ Mi) {
  const size_t idx = (size_t)blockIdx.x * 256 + threadIdx.x;  // over B*L*D
  const int d = (int)(idx & (D_ - 1));
  const size_t bl = idx >> 10;
  const int b = (int)(bl >> 10);
  const int l = (int)(bl & (L_ - 1));
  const int h = d >> 6;
  const size_t cs = (size_t)(b * H_ + h) * L_ + l;
  const float c = Cb[cs], s = Sb[cs];
  const float vr = Vr[idx], vi = Vi[idx];
  Mr[idx] = (bf16)(vr * c - vi * s);
  Mi[idx] = (bf16)(vr * s + vi * c);
}

// ---------------------------------------------------------------------------
// LayerNorm of (alpha*X + beta*Y) over D=1024, one block per row.
// Optional f32 and bf16 outputs.
// ---------------------------------------------------------------------------
__global__ __launch_bounds__(256) void ln_kernel(
    const float* __restrict__ X, const float* __restrict__ Y,
    float alpha, float beta,
    const float* __restrict__ g, const float* __restrict__ bb,
    float* __restrict__ outF, bf16* __restrict__ outB) {
  __shared__ float redS[8], redQ[8];
  const int row = blockIdx.x;
  const int tid = threadIdx.x;
  const float* x = X + (size_t)row * D_;
  const float* y = Y + (size_t)row * D_;

  float t[4];
  float sum = 0.f, sq = 0.f;
  for (int j = 0; j < 4; ++j) {
    const int c = tid + j * 256;
    const float v = alpha * x[c] + beta * y[c];
    t[j] = v;
    sum += v;
    sq += v * v;
  }
  for (int o = 16; o; o >>= 1) {
    sum += __shfl_xor(sum, o, 32);
    sq  += __shfl_xor(sq, o, 32);
  }
  const int lane = tid & 31, wave = tid >> 5;
  if (lane == 0) { redS[wave] = sum; redQ[wave] = sq; }
  __syncthreads();
  float ts = 0.f, tq = 0.f;
  for (int w = 0; w < 8; ++w) { ts += redS[w]; tq += redQ[w]; }
  const float mean = ts * (1.f / D_);
  const float var  = tq * (1.f / D_) - mean * mean;
  const float rstd = rsqrtf(var + EPS_);
  for (int j = 0; j < 4; ++j) {
    const int c = tid + j * 256;
    const float v = (t[j] - mean) * rstd * g[c] + bb[c];
    if (outF) outF[(size_t)row * D_ + c] = v;
    if (outB) outB[(size_t)row * D_ + c] = (bf16)v;
  }
}

// ---------------------------------------------------------------------------
// host orchestration
// ---------------------------------------------------------------------------
extern "C" void kernel_launch(void* const* d_in, const int* in_sizes, int n_in,
                              void* d_out, int out_size, void* d_ws, size_t ws_size,
                              hipStream_t stream) {
  (void)in_sizes; (void)n_in; (void)out_size; (void)ws_size;
  const float* r    = (const float*)d_in[0];
  const float* ii   = (const float*)d_in[1];
  const float* Wq   = (const float*)d_in[2];
  const float* bq   = (const float*)d_in[3];
  const float* Wk   = (const float*)d_in[4];
  const float* bk   = (const float*)d_in[5];
  const float* Wv   = (const float*)d_in[6];
  const float* bv   = (const float*)d_in[7];
  const float* Wo   = (const float*)d_in[8];
  const float* bo   = (const float*)d_in[9];
  const float* Wsem = (const float*)d_in[10];
  const float* Wsyn = (const float*)d_in[11];
  const float* Wrea = (const float*)d_in[12];
  const float* gr   = (const float*)d_in[13];
  const float* brv  = (const float*)d_in[14];
  const float* gi   = (const float*)d_in[15];
  const float* biv  = (const float*)d_in[16];

  char* ws = (char*)d_ws;
  size_t off = 0;
  auto take = [&](size_t bytes) -> void* {
    void* p = ws + off;
    off += (bytes + 255) & ~(size_t)255;
    return p;
  };
  bf16*  xrb = (bf16*)take((size_t)BLD_ * 2);
  bf16*  xib = (bf16*)take((size_t)BLD_ * 2);
  bf16*  wqb = (bf16*)take((size_t)D_ * D_ * 2);
  bf16*  wkb = (bf16*)take((size_t)D_ * D_ * 2);
  bf16*  wvb = (bf16*)take((size_t)D_ * D_ * 2);
  bf16*  wob = (bf16*)take((size_t)D_ * D_ * 2);
  bf16*  whb = (bf16*)take((size_t)D_ * D_ * 2);
  bf16*  Qrb = (bf16*)take((size_t)BLD_ * 2);   // } 16 MB contiguous -> Hi
  bf16*  Qib = (bf16*)take((size_t)BLD_ * 2);   // }
  bf16*  Krb = (bf16*)take((size_t)BLD_ * 2);   // } 16 MB contiguous -> Hr
  bf16*  Kib = (bf16*)take((size_t)BLD_ * 2);   // }
  float* Vr  = (float*)take((size_t)BLD_ * 4);  // -> ar after rotate
  float* Vi  = (float*)take((size_t)BLD_ * 4);  // -> ai after rotate
  float* Cb  = (float*)take((size_t)B_ * H_ * L_ * 4);
  float* Sb  = (float*)take((size_t)B_ * H_ * L_ * 4);
  bf16*  Mr  = (bf16*)take((size_t)BLD_ * 2);
  bf16*  Mi  = (bf16*)take((size_t)BLD_ * 2);
  float* r1  = (float*)take((size_t)BLD_ * 4);
  float* i1  = (float*)take((size_t)BLD_ * 4);
  bf16*  r1b = (bf16*)take((size_t)BLD_ * 2);
  bf16*  i1b = (bf16*)take((size_t)BLD_ * 2);

  // buffer aliases with disjoint lifetimes
  float* arf = Vr;          // Wo output (r stream) overwrites Vr
  float* aif = Vi;
  float* Hr  = (float*)Krb; // Kr+Ki region (16 MB) after attention is done
  float* Hi  = (float*)Qrb; // Qr+Qi region

  float* out = (float*)d_out;
  float* r2  = out;
  float* i2  = out + (size_t)BLD_;
  float* att = out + 2 * (size_t)BLD_;

  conv_weights_kernel<<<(D_ * D_) / 256, 256, 0, stream>>>(
      Wq, Wk, Wv, Wo, Wsem, Wsyn, Wrea, wqb, wkb, wvb, wob, whb);
  conv_inputs_kernel<<<BLD_ / 256, 256, 0, stream>>>(r, ii, xrb, xib);

  const dim3 gg(D_ / 64, (B_ * L_) / 128);
  gemm_xwT_bf16<<<gg, 256, 0, stream>>>(xrb, wqb, bq, nullptr, Qrb, B_ * L_, D_, D_);
  gemm_xwT_bf16<<<gg, 256, 0, stream>>>(xib, wqb, bq, nullptr, Qib, B_ * L_, D_, D_);
  gemm_xwT_bf16<<<gg, 256, 0, stream>>>(xrb, wkb, bk, nullptr, Krb, B_ * L_, D_, D_);
  gemm_xwT_bf16<<<gg, 256, 0, stream>>>(xib, wkb, bk, nullptr, Kib, B_ * L_, D_, D_);
  gemm_xwT_bf16<<<gg, 256, 0, stream>>>(xrb, wvb, bv, Vr, nullptr, B_ * L_, D_, D_);
  gemm_xwT_bf16<<<gg, 256, 0, stream>>>(xib, wvb, bv, Vi, nullptr, B_ * L_, D_, D_);

  (void)hipFuncSetAttribute((const void*)attn_kernel,
                            hipFuncAttributeMaxDynamicSharedMemorySize, 131072);
  attn_kernel<<<B_ * H_ * (L_ / 16), 256, 131072, stream>>>(
      Qrb, Qib, Krb, Kib, att, Cb, Sb);

  rotate_kernel<<<BLD_ / 256, 256, 0, stream>>>(Vr, Vi, Cb, Sb, Mr, Mi);

  gemm_xwT_bf16<<<gg, 256, 0, stream>>>(Mr, wob, bo, arf, nullptr, B_ * L_, D_, D_);
  gemm_xwT_bf16<<<gg, 256, 0, stream>>>(Mi, wob, bo, aif, nullptr, B_ * L_, D_, D_);

  ln_kernel<<<B_ * L_, 256, 0, stream>>>(r, arf, 1.f, 1.f, gr, brv, r1, r1b);
  ln_kernel<<<B_ * L_, 256, 0, stream>>>(ii, aif, 1.f, 1.f, gi, biv, i1, i1b);

  gemm_xwT_bf16<<<gg, 256, 0, stream>>>(r1b, whb, nullptr, Hr, nullptr, B_ * L_, D_, D_);
  gemm_xwT_bf16<<<gg, 256, 0, stream>>>(i1b, whb, nullptr, Hi, nullptr, B_ * L_, D_, D_);

  // r2 = LN(r1 + (r1 + DT*Hi)) = LN(2*r1 + DT*Hi); i2 = LN(2*i1 - DT*Hr)
  ln_kernel<<<B_ * L_, 256, 0, stream>>>(r1, Hi, 2.f, DT_, gr, brv, r2, nullptr);
  ln_kernel<<<B_ * L_, 256, 0, stream>>>(i1, Hr, 2.f, -DT_, gi, biv, i2, nullptr);
}
